// NearestEmbed_50611894616185
// MI455X (gfx1250) — matrix operand order repaired and checked by reading, not compiled, not run on hardware
//
#include <hip/hip_runtime.h>

typedef __attribute__((ext_vector_type(16))) _Float16 v16h;
typedef __attribute__((ext_vector_type(8)))  _Float16 v8h;
typedef __attribute__((ext_vector_type(8)))  float    v8f;
typedef __attribute__((ext_vector_type(4)))  int      v4i;

#define B_  64
#define D_  256
#define O_  1024
#define K_  1024
#define NT_ (K_ / 16)             // 64 N-tiles
#define TILE_BYTES (16 * D_ * 2)  // 8 KB per N-tile of f16 embT

#if __has_builtin(__builtin_amdgcn_global_load_async_to_lds_b128)
#define HAVE_ASYNC_LDS 1
#else
#define HAVE_ASYNC_LDS 0
#endif

__device__ __forceinline__ void wait_async0() {
#if __has_builtin(__builtin_amdgcn_s_wait_asynccnt)
  __builtin_amdgcn_s_wait_asynccnt(0);
#elif HAVE_ASYNC_LDS
  asm volatile("s_wait_asynccnt 0x0" ::: "memory");
#endif
}

// ---------------------------------------------------------------------------
// Kernel 1: e2[k] = sum_d emb[d][k]^2  AND  embT[k][d] = (f16)emb[d][k].
// embT is the B operand pre-transposed so the main loop stages contiguous
// 8 KB blocks per N-tile.
// ---------------------------------------------------------------------------
__global__ void prep_kernel(const float* __restrict__ emb,
                            float* __restrict__ e2,
                            _Float16* __restrict__ embT) {
  const int k = blockIdx.x * blockDim.x + threadIdx.x;   // 0..K-1
  float s = 0.f;
#pragma unroll 8
  for (int d = 0; d < D_; ++d) {
    const float v = emb[d * K_ + k];                     // coalesced across lanes
    s = __builtin_fmaf(v, v, s);
    embT[k * D_ + d] = (_Float16)v;                      // contiguous per thread
  }
  e2[k] = s;
}

// ---------------------------------------------------------------------------
// Kernel 2: fused GEMM (WMMA f16) + argmin epilogue.
// Block = 256 threads = 8 wave32; each wave owns 16 rows of O for one b.
// A tile (16 x 256 f16) in registers; B tiles async-copied global->LDS with
// double buffering so the copy of tile j+1 overlaps compute of tile j.
// ---------------------------------------------------------------------------
__global__ __launch_bounds__(256) void vq_argmin_kernel(
    const float* __restrict__ x, const _Float16* __restrict__ embT,
    const float* __restrict__ e2, float* __restrict__ argmin_f) {
  __shared__ __align__(32) _Float16 ldsB[2][16 * D_];    // 2 x 8 KB ping-pong

  const int lane = threadIdx.x & 31;
  const int wave = threadIdx.x >> 5;
  const int b    = blockIdx.x >> 3;                      // 8 blocks per batch entry
  const int o0   = (blockIdx.x & 7) * 128;               // 8 waves * 16 rows
  const int ow   = o0 + wave * 16;
  const int m    = lane & 15;                            // A row / B-C column N
  const int khiA = (lane < 16) ? 0 : 16;                 // A K split (contiguous, matches B)
  const int khiB = (lane < 16) ? 0 : 16;                 // B K split

  // ---- Stage helper: copy 8 KB tile j of embT into ldsB[buf].
  auto stage = [&](int j, int buf) {
    const char* gsrc = (const char*)embT + (size_t)j * TILE_BYTES;
    char*       ldst = (char*)&ldsB[buf][0];
    const int   off  = threadIdx.x * 16;                 // 256 threads x 16 B x 2
#if HAVE_ASYNC_LDS
    __builtin_amdgcn_global_load_async_to_lds_b128((v4i*)(gsrc + off),
                                                   (v4i*)(ldst + off), 0, 0);
    __builtin_amdgcn_global_load_async_to_lds_b128((v4i*)(gsrc + off + 4096),
                                                   (v4i*)(ldst + off + 4096), 0, 0);
#else
    *(v8h*)(ldst + off)        = *(const v8h*)(gsrc + off);
    *(v8h*)(ldst + off + 4096) = *(const v8h*)(gsrc + off + 4096);
#endif
  };

  // ---- Load this wave's A tile: A[m][k] = x[b][k][ow+m], f32 -> f16.
  const float* xg = x + (size_t)b * D_ * O_ + (ow + m);
  v16h a[8];
#pragma unroll
  for (int c8 = 0; c8 < 8; ++c8) {
    v16h av;
#pragma unroll
    for (int i = 0; i < 8; ++i) {
      av[i]     = (_Float16)xg[(c8 * 32 + khiA + i) * O_];
      av[i + 8] = (_Float16)xg[(c8 * 32 + khiA + 8 + i) * O_];
    }
    a[c8] = av;
  }

  float bestv[8];
  int   besti[8];
#pragma unroll
  for (int r = 0; r < 8; ++r) { bestv[r] = 3.4e38f; besti[r] = 0; }

  // Prologue: bring in tile 0.
  stage(0, 0);
  wait_async0();
  __syncthreads();

  for (int j = 0; j < NT_; ++j) {
    const int buf = j & 1;
    if (j + 1 < NT_) stage(j + 1, buf ^ 1);              // overlaps compute below

    // ---- 16x16 score tile over full D via 8 WMMAs (K=32 each).
    v8f c = {};
#pragma unroll
    for (int c8 = 0; c8 < 8; ++c8) {
      const v16h bb = *(const v16h*)(&ldsB[buf][m * D_ + c8 * 32 + khiB]);
      c = __builtin_amdgcn_wmma_f32_16x16x32_f16(
              false, a[c8], false, bb, (short)0, c, false, false);
    }

    // ---- Epilogue: dist = e2[n] - 2*score; per-lane running argmin.
    const int   ng  = j * 16 + m;
    const float e2v = e2[ng];
#pragma unroll
    for (int r = 0; r < 8; ++r) {
      const float dd = __builtin_fmaf(-2.0f, c[r], e2v);
      if (dd < bestv[r]) { bestv[r] = dd; besti[r] = ng; }
    }

    wait_async0();                                       // our async LDS writes done
    __syncthreads();                                     // publish next buffer
  }

  // ---- Cross-lane min within each 16-lane half (C layout: lanes 0-15 hold
  // M=r, lanes 16-31 hold M=r+8; N striped across the 16 lanes).
#pragma unroll
  for (int r = 0; r < 8; ++r) {
    float v  = bestv[r];
    int   ix = besti[r];
#pragma unroll
    for (int msk = 1; msk < 16; msk <<= 1) {
      const float ov = __shfl_xor(v,  msk, 32);
      const int   oi = __shfl_xor(ix, msk, 32);
      if (ov < v || (ov == v && oi < ix)) { v = ov; ix = oi; }  // ties -> low k
    }
    if ((lane & 15) == 0) {
      const int row = (lane == 0) ? r : r + 8;
      argmin_f[(size_t)b * O_ + ow + row] = (float)ix;   // exact for ix < 2^24
    }
  }
}

// ---------------------------------------------------------------------------
// Kernel 3: quant[b][d][o] = emb[d][argmin[b][o]]  (coalesced stores; emb in L2)
// ---------------------------------------------------------------------------
__global__ void gather_kernel(const float* __restrict__ emb,
                              const float* __restrict__ argmin_f,
                              float* __restrict__ quant) {
  const int t = blockIdx.x * blockDim.x + threadIdx.x;   // 0 .. B*O-1
  const int b = t >> 10;                                 // / O_
  const int o = t & (O_ - 1);
  const int k = (int)argmin_f[t];
  float* outp = quant + (size_t)b * D_ * O_ + o;
#pragma unroll 4
  for (int d = 0; d < D_; ++d)
    outp[d * O_] = emb[d * K_ + k];                      // stores coalesced over o
}

// ---------------------------------------------------------------------------
extern "C" void kernel_launch(void* const* d_in, const int* in_sizes, int n_in,
                              void* d_out, int out_size, void* d_ws, size_t ws_size,
                              hipStream_t stream) {
  (void)in_sizes; (void)n_in; (void)out_size; (void)ws_size;
  const float* x   = (const float*)d_in[0];   // (B, D, O) f32
  const float* emb = (const float*)d_in[1];   // (D, K)    f32

  float* quant    = (float*)d_out;                        // (B, D, O)
  float* argmin_f = quant + (size_t)B_ * D_ * O_;         // (B, O) as float values

  float*    e2   = (float*)d_ws;                          // 4 KB
  _Float16* embT = (_Float16*)((char*)d_ws + 4096);       // 512 KB, [k][d] f16

  prep_kernel<<<K_ / 256, 256, 0, stream>>>(emb, e2, embT);
  vq_argmin_kernel<<<(B_ * O_) / 128, 256, 0, stream>>>(x, embT, e2, argmin_f);
  gather_kernel<<<(B_ * O_) / 256, 256, 0, stream>>>(emb, argmin_f, quant);
}
